// SelfAttention_3728031613582
// MI455X (gfx1250) — matrix operand order repaired and checked by reading, not compiled, not run on hardware
//
#include <hip/hip_runtime.h>
#include <hip/hip_bf16.h>

// Problem constants (match reference)
#define S_LEN 2048
#define HID   4096
#define NH    32
#define HD    128
#define N3    12288   // 3*HID

typedef __attribute__((ext_vector_type(16))) _Float16 v16h;
typedef __attribute__((ext_vector_type(8)))  _Float16 v8h;
typedef __attribute__((ext_vector_type(8)))  float    v8f;

static __device__ inline v16h cat8(v8h a, v8h b) {
    return __builtin_shufflevector(a, b, 0,1,2,3,4,5,6,7,8,9,10,11,12,13,14,15);
}
static __device__ inline v8f wmma_f16(v16h a, v16h b, v8f c) {
    return __builtin_amdgcn_wmma_f32_16x16x32_f16(false, a, false, b, (short)0, c, false, false);
}
// A fragment (16x32 f16): lane r=lane&15 -> row, hf=lane>>4 -> k in {hf*8..+7} u {hf*8+16..+23}
static __device__ inline v16h ld_afrag(const _Float16* p) {
    v8h lo = *(const v8h*)(p);
    v8h hi = *(const v8h*)(p + 16);
    return cat8(lo, hi);
}
// B fragment (32x16 f16): lane r -> col n, hf -> k in {hf*8..+7} u {hf*8+16..+23}
static __device__ inline v16h ld_bfrag(const _Float16* p) {
    v8h lo = *(const v8h*)(p);
    v8h hi = *(const v8h*)(p + 16);
    return cat8(lo, hi);
}
static __device__ inline v8h pk2(float4 a, float4 b) {
    return (v8h){(_Float16)a.x,(_Float16)a.y,(_Float16)a.z,(_Float16)a.w,
                 (_Float16)b.x,(_Float16)b.y,(_Float16)b.z,(_Float16)b.w};
}
// low 32 bits of a flat shared pointer == LDS byte offset
static __device__ inline unsigned lds_off(const void* p) {
    return (unsigned)(unsigned long long)p;
}
// gfx1250 async global->LDS copy (16B per lane), tracked by ASYNCcnt
static __device__ inline void async_g2l_b128(unsigned ldsAddr, const void* gaddr) {
    asm volatile("global_load_async_to_lds_b128 %0, %1, off"
                 :: "v"(ldsAddr), "v"(gaddr) : "memory");
}
static __device__ inline void wait_async0() {
    asm volatile("s_wait_asynccnt 0x0" ::: "memory");
}

// ---------------------------------------------------------------------------
// Kernel 1: QKV GEMM  (M=2048, N=12288, K=4096), f32 in -> f16 tiles -> WMMA
// Double-buffered LDS; epilogue +bias, scatter to Qh/Kh [NH][S][HD], VTh [NH][HD][S]
// ---------------------------------------------------------------------------
#define BM 128
#define BN 128
#define BK 32
#define LDT 40   // padded LDS row stride (halves): conflict-free b128 frag loads

__global__ __launch_bounds__(256) void qkv_gemm_kernel(
    const float* __restrict__ hid, const float* __restrict__ wqkv,
    const float* __restrict__ bqkv,
    _Float16* __restrict__ Qh, _Float16* __restrict__ Kh, _Float16* __restrict__ VTh)
{
    __shared__ _Float16 As[2][BM * LDT];
    __shared__ _Float16 Bs[2][BM * LDT];

    const int nBase = blockIdx.x * BN;
    const int mBase = blockIdx.y * BM;
    const int tid  = threadIdx.x;
    const int lane = tid & 31, wid = tid >> 5;
    const int waveM = wid >> 2;      // 0..1  -> 64 rows each
    const int waveN = wid & 3;       // 0..3  -> 32 cols each
    const int r = lane & 15, hf = lane >> 4;

    v8f acc[4][2];
#pragma unroll
    for (int i = 0; i < 4; ++i)
#pragma unroll
        for (int j = 0; j < 2; ++j)
            acc[i][j] = (v8f){0.f,0.f,0.f,0.f,0.f,0.f,0.f,0.f};

    const int ldRow = tid >> 1;          // 0..127
    const int ldCol = (tid & 1) * 16;    // 0 or 16
    const float* aG = hid  + (size_t)(mBase + ldRow) * HID + ldCol;
    const float* bG = wqkv + (size_t)(nBase + ldRow) * HID + ldCol;
    const int lo = ldRow * LDT + ldCol;

    float4 ra0, ra1, ra2, ra3, rb0, rb1, rb2, rb3;
    auto gload = [&](int k0) {
        ra0 = *(const float4*)(aG + k0);     ra1 = *(const float4*)(aG + k0 + 4);
        ra2 = *(const float4*)(aG + k0 + 8); ra3 = *(const float4*)(aG + k0 + 12);
        rb0 = *(const float4*)(bG + k0);     rb1 = *(const float4*)(bG + k0 + 4);
        rb2 = *(const float4*)(bG + k0 + 8); rb3 = *(const float4*)(bG + k0 + 12);
    };
    auto lstore = [&](int buf) {
        *(v8h*)(&As[buf][lo])     = pk2(ra0, ra1);
        *(v8h*)(&As[buf][lo + 8]) = pk2(ra2, ra3);
        *(v8h*)(&Bs[buf][lo])     = pk2(rb0, rb1);
        *(v8h*)(&Bs[buf][lo + 8]) = pk2(rb2, rb3);
    };

    gload(0); lstore(0);
    __syncthreads();
    const int NK = HID / BK;
    for (int kt = 0; kt < NK; ++kt) {
        const int cur = kt & 1;
        if (kt + 1 < NK) gload((kt + 1) * BK);
        v16h af[4], bf[2];
#pragma unroll
        for (int i = 0; i < 4; ++i)
            af[i] = ld_afrag(&As[cur][(waveM*64 + i*16 + r)*LDT + hf*8]);
#pragma unroll
        for (int j = 0; j < 2; ++j)
            bf[j] = ld_bfrag(&Bs[cur][(waveN*32 + j*16 + r)*LDT + hf*8]);
#pragma unroll
        for (int i = 0; i < 4; ++i)
#pragma unroll
            for (int j = 0; j < 2; ++j)
                acc[i][j] = wmma_f16(af[i], bf[j], acc[i][j]);
        if (kt + 1 < NK) lstore(cur ^ 1);
        __syncthreads();
    }

    // epilogue: each 128-wide N block is exactly one (head, type) since 384 = 3*128
    const int head = nBase / 384;
    const int typ  = (nBase % 384) >> 7;     // 0=q, 1=k, 2=v  (block-uniform)
    float bias[2];
#pragma unroll
    for (int j = 0; j < 2; ++j) bias[j] = bqkv[nBase + waveN*32 + j*16 + r];

    if (typ < 2) {
        _Float16* D = (typ == 0 ? Qh : Kh) + (size_t)head * S_LEN * HD;
#pragma unroll
        for (int i = 0; i < 4; ++i)
#pragma unroll
            for (int j = 0; j < 2; ++j)
#pragma unroll
                for (int e = 0; e < 8; ++e) {
                    int m  = mBase + waveM*64 + i*16 + e + hf*8;
                    int nc = waveN*32 + j*16 + r;
                    D[(size_t)m * HD + nc] = (_Float16)(acc[i][j][e] + bias[j]);
                }
    } else {
        _Float16* D = VTh + (size_t)head * HD * S_LEN;
#pragma unroll
        for (int i = 0; i < 4; ++i)
#pragma unroll
            for (int j = 0; j < 2; ++j)
#pragma unroll
                for (int e = 0; e < 8; ++e) {
                    int m  = mBase + waveM*64 + i*16 + e + hf*8;
                    int nc = waveN*32 + j*16 + r;
                    D[(size_t)nc * S_LEN + m] = (_Float16)(acc[i][j][e] + bias[j]);
                }
    }
}

// ---------------------------------------------------------------------------
// Kernel 2: 2D RoPE in place on Qh/Kh (f32 math); folds 1/sqrt(HD) into Q.
// coeff cancels: softmax((qk*coeff)/(sqrt(HD)*coeff)) == softmax(qk/sqrt(HD))
// ---------------------------------------------------------------------------
__global__ __launch_bounds__(128) void rope_kernel(
    const int* __restrict__ pid, _Float16* __restrict__ Qh, _Float16* __restrict__ Kh)
{
    const int s    = blockIdx.x;
    const int head = blockIdx.y;
    const int d    = threadIdx.x;        // 0..127
    const int h2 = d >> 6, dd = d & 63, fi = dd & 31;
    const int p = (h2 == 0) ? pid[s] : pid[S_LEN + s];   // position_ids[0,{0,1},s]
    const float inv = __expf((float)fi * (-0.2878231366242557f)); // 10000^(-fi/32)
    float sn, cs;
    sincosf((float)p * inv, &sn, &cs);
    const size_t base = ((size_t)head * S_LEN + s) * HD;
    const int partner = h2*64 + ((dd < 32) ? (dd + 32) : (dd - 32));
    const float sgn = (dd < 32) ? -1.f : 1.f;
    float qx = (float)Qh[base + d], qp = (float)Qh[base + partner];
    float kx = (float)Kh[base + d], kp = (float)Kh[base + partner];
    __syncthreads();   // partners read before anyone writes
    const float rs = 0.08838834764831845f;   // 1/sqrt(128)
    Qh[base + d] = (_Float16)((qx*cs + sgn*qp*sn) * rs);
    Kh[base + d] = (_Float16)( kx*cs + sgn*kp*sn);
}

// ---------------------------------------------------------------------------
// Kernel 3: causal flash attention per (head, 128-row Q block).
// 8 waves x 16 rows; 64-key KV tiles loaded with async global->LDS copies;
// QK^T and P.V via WMMA f16->f32.
// ---------------------------------------------------------------------------
#define KVT 64
#define LDK 136   // sK row stride (halves), conflict-free
#define LDV 72    // sV / sP row stride (halves), conflict-free

__global__ __launch_bounds__(256) void attn_kernel(
    const _Float16* __restrict__ Qh, const _Float16* __restrict__ Kh,
    const _Float16* __restrict__ VTh, _Float16* __restrict__ ctx)
{
    __shared__ _Float16 sK[KVT * LDK];       // [key][hd]
    __shared__ _Float16 sV[HD * LDV];        // [hd][key]  (from VTh)
    __shared__ _Float16 sP[8 * 16 * LDV];    // per-wave 16x64 P tile

    const int head = blockIdx.x;
    const int q0   = blockIdx.y * 128;
    const int tid = threadIdx.x, lane = tid & 31, wid = tid >> 5;
    const int r = lane & 15, hf = lane >> 4;
    const size_t hQ = (size_t)head * S_LEN * HD;   // Qh/Kh head base
    const size_t hV = (size_t)head * HD * S_LEN;   // VTh head base

    // Q fragments for this wave's 16 rows (A layout, direct from global)
    v16h qf[4];
    {
        const _Float16* qp = Qh + hQ + (size_t)(q0 + wid*16 + r)*HD + hf*8;
#pragma unroll
        for (int kc = 0; kc < 4; ++kc)
            qf[kc] = cat8(*(const v8h*)(qp + kc*32), *(const v8h*)(qp + kc*32 + 16));
    }

    v8f acc[8];
#pragma unroll
    for (int nf = 0; nf < 8; ++nf) acc[nf] = (v8f){0.f,0.f,0.f,0.f,0.f,0.f,0.f,0.f};
    float mrow[8], lrow[8];
#pragma unroll
    for (int e = 0; e < 8; ++e) { mrow[e] = -3.0e38f; lrow[e] = 0.f; }

    const int rowg0 = q0 + wid*16 + hf*8;
    _Float16* pw = sP + wid * 16 * LDV;

    // per-thread tile-copy coordinates
    const int kRow = tid >> 2, kSeg = (tid & 3) * 32;   // K tile: 64 x 128
    const int vRow = tid >> 1, vSeg = (tid & 1) * 32;   // V tile: 128 x 64
    const unsigned lK = lds_off(sK + kRow*LDK + kSeg);
    const unsigned lV = lds_off(sV + vRow*LDV + vSeg);

    for (int kv = 0; kv < q0 + 128; kv += KVT) {
        { // async copy K tile [64][128] and V^T tile [128][64] into LDS
            const _Float16* gk = Kh  + hQ + (size_t)(kv + kRow)*HD + kSeg;
            const _Float16* gv = VTh + hV + (size_t)vRow * S_LEN + kv + vSeg;
#pragma unroll
            for (int c = 0; c < 4; ++c) async_g2l_b128(lK + c*16, gk + c*8);
#pragma unroll
            for (int c = 0; c < 4; ++c) async_g2l_b128(lV + c*16, gv + c*8);
            wait_async0();
        }
        __syncthreads();

        // scores: 16 rows x 64 keys
        v8f sc[4];
#pragma unroll
        for (int j = 0; j < 4; ++j) sc[j] = (v8f){0.f,0.f,0.f,0.f,0.f,0.f,0.f,0.f};
#pragma unroll
        for (int kc = 0; kc < 4; ++kc)
#pragma unroll
            for (int j = 0; j < 4; ++j) {
                v16h b = ld_bfrag(sK + (j*16 + r)*LDK + kc*32 + hf*8);
                sc[j] = wmma_f16(qf[kc], b, sc[j]);
            }

        // causal mask + online softmax
        float mnew[8];
#pragma unroll
        for (int e = 0; e < 8; ++e) {
            const int rowg = rowg0 + e;
            float mx = -3.0e38f;
#pragma unroll
            for (int j = 0; j < 4; ++j) {
                float v = sc[j][e];
                if (kv + j*16 + r > rowg) v = -3.0e38f;
                sc[j][e] = v;
                mx = fmaxf(mx, v);
            }
            mx = fmaxf(mx, __shfl_xor(mx, 1, 32));
            mx = fmaxf(mx, __shfl_xor(mx, 2, 32));
            mx = fmaxf(mx, __shfl_xor(mx, 4, 32));
            mx = fmaxf(mx, __shfl_xor(mx, 8, 32));
            mnew[e] = fmaxf(mrow[e], mx);
        }
#pragma unroll
        for (int e = 0; e < 8; ++e) {
            float scale = __expf(mrow[e] - mnew[e]);
            mrow[e] = mnew[e];
            lrow[e] *= scale;
#pragma unroll
            for (int nf = 0; nf < 8; ++nf) acc[nf][e] *= scale;
            float ssum = 0.f;
#pragma unroll
            for (int j = 0; j < 4; ++j) {
                float pv = __expf(sc[j][e] - mnew[e]);
                ssum += pv;
                pw[(e + hf*8)*LDV + j*16 + r] = (_Float16)pv;  // P row-major
            }
            ssum += __shfl_xor(ssum, 1, 32);
            ssum += __shfl_xor(ssum, 2, 32);
            ssum += __shfl_xor(ssum, 4, 32);
            ssum += __shfl_xor(ssum, 8, 32);
            lrow[e] += ssum;
        }
        asm volatile("s_wait_dscnt 0x0" ::: "memory"); // wave-local P RAW (LDS in-order per wave)

        // context += P (16x64) . V (64x128)
#pragma unroll
        for (int kc = 0; kc < 2; ++kc) {
            const _Float16* ap = pw + r*LDV + kc*32 + hf*8;
            v16h aP = cat8(*(const v8h*)ap, *(const v8h*)(ap + 16));
#pragma unroll
            for (int nf = 0; nf < 8; ++nf) {
                v16h bV = ld_bfrag(sV + (nf*16 + r)*LDV + kc*32 + hf*8);
                acc[nf] = wmma_f16(aP, bV, acc[nf]);
            }
        }
        __syncthreads();
    }

    // normalize + store context [S][NH*HD] as f16
#pragma unroll
    for (int e = 0; e < 8; ++e) {
        const float invl = 1.f / lrow[e];
        const int sg = rowg0 + e;
#pragma unroll
        for (int nf = 0; nf < 8; ++nf)
            ctx[(size_t)sg*HID + head*HD + nf*16 + r] = (_Float16)(acc[nf][e] * invl);
    }
}

// ---------------------------------------------------------------------------
// Kernel 4: dense output GEMM (M=2048, N=4096, K=4096) + bias -> f32 out
// ---------------------------------------------------------------------------
__global__ __launch_bounds__(256) void dense_gemm_kernel(
    const _Float16* __restrict__ ctx, const float* __restrict__ wd,
    const float* __restrict__ bd, float* __restrict__ out)
{
    __shared__ _Float16 As[2][BM * LDT];
    __shared__ _Float16 Bs[2][BM * LDT];

    const int nBase = blockIdx.x * BN;
    const int mBase = blockIdx.y * BM;
    const int tid = threadIdx.x;
    const int lane = tid & 31, wid = tid >> 5;
    const int waveM = wid >> 2, waveN = wid & 3;
    const int r = lane & 15, hf = lane >> 4;

    v8f acc[4][2];
#pragma unroll
    for (int i = 0; i < 4; ++i)
#pragma unroll
        for (int j = 0; j < 2; ++j)
            acc[i][j] = (v8f){0.f,0.f,0.f,0.f,0.f,0.f,0.f,0.f};

    const int ldRow = tid >> 1;
    const int ldCol = (tid & 1) * 16;
    const _Float16* aG = ctx + (size_t)(mBase + ldRow) * HID + ldCol;
    const float*    bG = wd  + (size_t)(nBase + ldRow) * HID + ldCol;
    const int lo = ldRow * LDT + ldCol;

    v8h ra0, ra1;
    float4 rb0, rb1, rb2, rb3;
    auto gload = [&](int k0) {
        ra0 = *(const v8h*)(aG + k0);
        ra1 = *(const v8h*)(aG + k0 + 8);
        rb0 = *(const float4*)(bG + k0);     rb1 = *(const float4*)(bG + k0 + 4);
        rb2 = *(const float4*)(bG + k0 + 8); rb3 = *(const float4*)(bG + k0 + 12);
    };
    auto lstore = [&](int buf) {
        *(v8h*)(&As[buf][lo])     = ra0;
        *(v8h*)(&As[buf][lo + 8]) = ra1;
        *(v8h*)(&Bs[buf][lo])     = pk2(rb0, rb1);
        *(v8h*)(&Bs[buf][lo + 8]) = pk2(rb2, rb3);
    };

    gload(0); lstore(0);
    __syncthreads();
    const int NK = HID / BK;
    for (int kt = 0; kt < NK; ++kt) {
        const int cur = kt & 1;
        if (kt + 1 < NK) gload((kt + 1) * BK);
        v16h af[4], bf[2];
#pragma unroll
        for (int i = 0; i < 4; ++i)
            af[i] = ld_afrag(&As[cur][(waveM*64 + i*16 + r)*LDT + hf*8]);
#pragma unroll
        for (int j = 0; j < 2; ++j)
            bf[j] = ld_bfrag(&Bs[cur][(waveN*32 + j*16 + r)*LDT + hf*8]);
#pragma unroll
        for (int i = 0; i < 4; ++i)
#pragma unroll
            for (int j = 0; j < 2; ++j)
                acc[i][j] = wmma_f16(af[i], bf[j], acc[i][j]);
        if (kt + 1 < NK) lstore(cur ^ 1);
        __syncthreads();
    }

    const float biasv[2] = { bd[nBase + waveN*32 + r],
                             bd[nBase + waveN*32 + 16 + r] };
#pragma unroll
    for (int i = 0; i < 4; ++i)
#pragma unroll
        for (int j = 0; j < 2; ++j)
#pragma unroll
            for (int e = 0; e < 8; ++e) {
                int m  = mBase + waveM*64 + i*16 + e + hf*8;
                int nc = nBase + waveN*32 + j*16 + r;
                out[(size_t)m*HID + nc] = acc[i][j][e] + biasv[j];
            }
}

// ---------------------------------------------------------------------------
extern "C" void kernel_launch(void* const* d_in, const int* in_sizes, int n_in,
                              void* d_out, int out_size, void* d_ws, size_t ws_size,
                              hipStream_t stream) {
    (void)in_sizes; (void)n_in; (void)out_size; (void)ws_size;
    const float* hid  = (const float*)d_in[0];
    const int*   pid  = (const int*)  d_in[1];  // position_ids [1,2,S]
    // d_in[2] attention_mask: causal triu -> hardcoded
    const float* wqkv = (const float*)d_in[3];
    const float* bqkv = (const float*)d_in[4];
    const float* wd   = (const float*)d_in[5];
    const float* bd   = (const float*)d_in[6];
    // d_in[7] layer_id: coeff cancels in softmax
    float* out = (float*)d_out;

    // workspace (f16): Qh | Kh | VTh | ctx  = 4 * 16MB = 64MB
    _Float16* ws  = (_Float16*)d_ws;
    const size_t seg = (size_t)NH * S_LEN * HD;   // 8M halves
    _Float16* Qh  = ws;
    _Float16* Kh  = ws + seg;
    _Float16* VTh = ws + 2*seg;
    _Float16* ctx = ws + 3*seg;

    qkv_gemm_kernel<<<dim3(N3/BN, S_LEN/BM), 256, 0, stream>>>(hid, wqkv, bqkv, Qh, Kh, VTh);
    rope_kernel<<<dim3(S_LEN, NH), 128, 0, stream>>>(pid, Qh, Kh);
    attn_kernel<<<dim3(NH, S_LEN/128), 256, 0, stream>>>(Qh, Kh, VTh, ctx);
    dense_gemm_kernel<<<dim3(HID/BN, S_LEN/BM), 256, 0, stream>>>(ctx, wd, bd, out);
}
